// AttnRNNCell_46755013984624
// MI455X (gfx1250) — matrix operand order repaired and checked by reading, not compiled, not run on hardware
//
#include <hip/hip_runtime.h>
#include <cstdint>
#include <cstddef>

// ---------------------------------------------------------------------------
// AttnRNNCell fused pipeline for MI455X (gfx1250, wave32, WMMA)
//   B=8192, IN=1024, H=2048, NH=4, HD=512
// GEMM-compute-bound (~309 GFLOP vs ~15us HBM traffic) -> bf16 WMMA path.
// v4: disable compiler re-unrolling of the ping-pong K loop (v3's x2 unroll
//     blew register pressure -> accumulator v_mov shuffling); 32-bit per-lane
//     offsets off uniform bases so loads use scalar-base + voffset addressing
//     instead of per-lane 64-bit adds.
// ---------------------------------------------------------------------------

typedef __bf16 bf16_t;
typedef __attribute__((ext_vector_type(16))) __bf16 v16bf;
typedef __attribute__((ext_vector_type(8)))  __bf16 v8bf;
typedef __attribute__((ext_vector_type(8)))  float  v8f;

#define B_   8192
#define IN_  1024
#define H_   2048
#define NH_  4
#define HD_  512
#define EPS_ 1e-5f

// ---------------------------------------------------------------- convert ---
__global__ __launch_bounds__(256)
void cvt_f32_bf16(const float* __restrict__ in, bf16_t* __restrict__ out, int n) {
  int i = blockIdx.x * blockDim.x + threadIdx.x;
  const int stride = gridDim.x * blockDim.x;
  for (; i < n; i += stride) out[i] = (bf16_t)in[i];
}

// ------------------------------------------------------------------- GEMM ---
// out[m,n] = act( sum_k A(m,k) * W[n,k] + bias[n] ),  N = H_ = 2048
// A(m,k) = A[m*LDA + k]          for k <  K1
//        = A2[m*LDA2 + (k-K1)]   for k >= K1   (concat along K)
// W is [N x KTOT] row-major (K contiguous) -> matches X @ W^T.
// Wave tile: 32(M) x 64(N) = 2 A-frags x 4 B-frags -> 8 accumulators.
// Block = 8 waves arranged 4(M) x 2(N) -> 128x128 block tile.
// K-loop: ping-pong double buffer, 64 K per iteration, last chunk peeled,
//         compiler unrolling disabled (register pressure control).
// ACT: 0 = identity, 1 = exact gelu, 2 = sigmoid.
template<int ACT, int K1, int KTOT, int LDA, int LDA2>
__global__ __launch_bounds__(256)
void gemm_bf16_wmma(const bf16_t* __restrict__ A,
                    const bf16_t* __restrict__ A2,
                    const bf16_t* __restrict__ W,
                    const float*  __restrict__ bias,
                    float* __restrict__ outF, bf16_t* __restrict__ outB)
{
  static_assert(KTOT % 64 == 0 && K1 % 64 == 0, "K must be a multiple of 64");
  constexpr int N = H_;

  const int lane = threadIdx.x & 31;
  const int wave = threadIdx.x >> 5;
  const int wm   = wave >> 1;                 // 0..3
  const int wn   = wave & 1;                  // 0..1
  const int mbase = blockIdx.y * 128 + wm * 32;
  const int nbase = blockIdx.x * 128 + wn * 64;
  const int half = lane >> 4;                 // lane group (0: lanes 0-15)
  const int l16  = lane & 15;

  // A-fragment (contiguous k order, matching B): lane m = l16,
  // K = e + 16*half -> two 16B chunks.
  // 32-bit per-lane element offsets -> scalar-base + voffset addressing.
  auto loadA = [&](int kt, int mi) -> v16bf {
    const bf16_t* Ab;
    uint32_t kk, ld;
    if (kt < K1) { Ab = A;  kk = (uint32_t)kt;        ld = (uint32_t)LDA;  }
    else         { Ab = A2; kk = (uint32_t)(kt - K1); ld = (uint32_t)LDA2; }
    const uint32_t mrow = (uint32_t)(mbase + mi * 16 + l16);
    const uint32_t o = mrow * ld + kk + (uint32_t)(16 * half);
    v8bf a0 = *(const v8bf*)(Ab + o);
    v8bf a1 = *(const v8bf*)(Ab + o + 8);
    __builtin_prefetch(Ab + o + 32, 0, 3);    // near-scope global_prefetch_b8
    return __builtin_shufflevector(a0, a1,
        0, 1, 2, 3, 4, 5, 6, 7, 8, 9, 10, 11, 12, 13, 14, 15);
  };
  // B-fragment (contiguous k order, matching A): lane n = l16, K = e + 16*half
  // -> one contiguous 32B load per lane.
  auto loadB = [&](int kt, int nt) -> v16bf {
    const uint32_t o = (uint32_t)(nbase + nt * 16 + l16) * (uint32_t)KTOT
                     + (uint32_t)(kt + 16 * half);
    return *(const v16bf*)(W + o);
  };

  v8f acc[2][4] = {};
  v16bf A0[2], B0[4], A1[2], B1[4];

  auto loadStep = [&](int kt, v16bf (&Af)[2], v16bf (&Bf)[4]) {
#pragma unroll
    for (int mi = 0; mi < 2; ++mi) Af[mi] = loadA(kt, mi);
#pragma unroll
    for (int nt = 0; nt < 4; ++nt) Bf[nt] = loadB(kt, nt);
  };
  auto mmaStep = [&](v16bf (&Af)[2], v16bf (&Bf)[4]) {
#pragma unroll
    for (int mi = 0; mi < 2; ++mi)
#pragma unroll
      for (int nt = 0; nt < 4; ++nt)
        acc[mi][nt] = __builtin_amdgcn_wmma_f32_16x16x32_bf16(
            false, Af[mi], false, Bf[nt], (short)0, acc[mi][nt],
            false, false);
  };

  // Software pipeline: load(kt+32) ahead of mma(kt); ping-pong buffers.
  loadStep(0, A0, B0);
  int kt = 0;
#pragma clang loop unroll(disable)
  for (; kt + 64 < KTOT; kt += 64) {
    loadStep(kt + 32, A1, B1);
    mmaStep(A0, B0);
    loadStep(kt + 64, A0, B0);
    mmaStep(A1, B1);
  }
  // Final 64-K chunk (kt == KTOT - 64): no out-of-range loads.
  loadStep(kt + 32, A1, B1);
  mmaStep(A0, B0);
  mmaStep(A1, B1);

  // Epilogue. C/D layout: VGPR r -> row M = r + 8*half, col N = l16.
#pragma unroll
  for (int mi = 0; mi < 2; ++mi) {
#pragma unroll
    for (int nt = 0; nt < 4; ++nt) {
      const int n = nbase + nt * 16 + l16;
      const float bn = bias ? bias[n] : 0.0f;
#pragma unroll
      for (int r = 0; r < 8; ++r) {
        const int m = mbase + mi * 16 + r + 8 * half;
        float v = acc[mi][nt][r] + bn;
        if (ACT == 1) v = 0.5f * v * (1.0f + erff(v * 0.70710678118654752440f));
        if (ACT == 2) v = 1.0f / (1.0f + __expf(-v));
        const uint32_t idx = (uint32_t)m * (uint32_t)N + (uint32_t)n;
        if (outF) outF[idx] = v;
        if (outB) outB[idx] = (bf16_t)v;
      }
    }
  }
}

// -------------------------------------------------------------- attention ---
// One wave per (batch b, head). 4 keys: [h, xp, h+xp, h*xp]; q.k / sqrt(HD);
// softmax over 4; weighted sum -> attn (stored bf16 for the W_o GEMM).
__global__ __launch_bounds__(128)
void attn_kernel(const float* __restrict__ hprev, const float* __restrict__ xproj,
                 const float* __restrict__ q, bf16_t* __restrict__ attnb)
{
  const int b    = blockIdx.x;
  const int head = threadIdx.x >> 5;
  const int lane = threadIdx.x & 31;
  const uint32_t base = (uint32_t)b * H_ + (uint32_t)head * HD_;

  float hv[16], xv[16];
  float s0 = 0.f, s1 = 0.f, s2 = 0.f, s3 = 0.f;
#pragma unroll
  for (int e = 0; e < 16; ++e) {
    const uint32_t d = base + (uint32_t)(e * 32 + lane);
    const float qq = q[d];
    const float h  = hprev[d];
    const float x  = xproj[d];
    hv[e] = h; xv[e] = x;
    s0 += qq * h;
    s1 += qq * x;
    s2 += qq * (h + x);
    s3 += qq * (h * x);
  }
#pragma unroll
  for (int off = 16; off > 0; off >>= 1) {
    s0 += __shfl_xor(s0, off, 32);
    s1 += __shfl_xor(s1, off, 32);
    s2 += __shfl_xor(s2, off, 32);
    s3 += __shfl_xor(s3, off, 32);
  }
  const float scale = 0.04419417382415922f;   // 1/sqrt(512)
  s0 *= scale; s1 *= scale; s2 *= scale; s3 *= scale;
  const float mx = fmaxf(fmaxf(s0, s1), fmaxf(s2, s3));
  float w0 = __expf(s0 - mx), w1 = __expf(s1 - mx);
  float w2 = __expf(s2 - mx), w3 = __expf(s3 - mx);
  const float inv = 1.0f / (w0 + w1 + w2 + w3);
  w0 *= inv; w1 *= inv; w2 *= inv; w3 *= inv;
#pragma unroll
  for (int e = 0; e < 16; ++e) {
    const uint32_t d = base + (uint32_t)(e * 32 + lane);
    const float h = hv[e], x = xv[e];
    const float a = w0 * h + w1 * x + w2 * (h + x) + w3 * (h * x);
    attnb[d] = (bf16_t)a;
  }
}

// ----------------------------------------------------- layernorm + gating ---
__global__ __launch_bounds__(256)
void final_kernel(const float* __restrict__ hprev, const float* __restrict__ og,
                  const float* __restrict__ gate, const float* __restrict__ gamma,
                  const float* __restrict__ beta, float* __restrict__ out)
{
  const int b = blockIdx.x;
  const int t = threadIdx.x;
  const uint32_t base = (uint32_t)b * H_;

  float y[8], h[8];
  float s = 0.f, sq = 0.f;
#pragma unroll
  for (int e = 0; e < 8; ++e) {
    const uint32_t n = base + (uint32_t)(e * 256 + t);
    const float hh = hprev[n];
    const float yy = og[n] + hh;
    h[e] = hh; y[e] = yy;
    s += yy; sq += yy * yy;
  }
  __shared__ float rs[256];
  __shared__ float rq[256];
  rs[t] = s; rq[t] = sq;
  __syncthreads();
  for (int off = 128; off > 0; off >>= 1) {
    if (t < off) { rs[t] += rs[t + off]; rq[t] += rq[t + off]; }
    __syncthreads();
  }
  const float mean = rs[0] * (1.0f / H_);
  const float var  = rq[0] * (1.0f / H_) - mean * mean;
  const float rstd = rsqrtf(var + EPS_);
#pragma unroll
  for (int e = 0; e < 8; ++e) {
    const int n = e * 256 + t;
    const float cand = (y[e] - mean) * rstd * gamma[n] + beta[n];
    const float g = gate[base + n];
    out[base + n] = g * cand + (1.0f - g) * h[e];
  }
}

// ----------------------------------------------------------------- launch ---
extern "C" void kernel_launch(void* const* d_in, const int* in_sizes, int n_in,
                              void* d_out, int out_size, void* d_ws, size_t ws_size,
                              hipStream_t stream)
{
  const float* h_prev = (const float*)d_in[0];
  const float* x      = (const float*)d_in[1];
  const float* W_proj = (const float*)d_in[2];
  const float* b_proj = (const float*)d_in[3];
  const float* W_q    = (const float*)d_in[4];
  const float* b_q    = (const float*)d_in[5];
  const float* W_o    = (const float*)d_in[6];
  const float* b_o    = (const float*)d_in[7];
  const float* W_g    = (const float*)d_in[8];
  const float* b_g    = (const float*)d_in[9];
  const float* gamma  = (const float*)d_in[10];
  const float* beta   = (const float*)d_in[11];
  float* out = (float*)d_out;

  char* ws = (char*)d_ws;
  size_t off = 0;
  auto walloc = [&](size_t bytes) -> char* {
    char* p = ws + off;
    off += (bytes + 255) & ~(size_t)255;
    return p;
  };

  bf16_t* xb    = (bf16_t*)walloc((size_t)B_ * IN_ * 2);      // x, bf16
  bf16_t* hb    = (bf16_t*)walloc((size_t)B_ * H_ * 2);       // h_prev, bf16
  bf16_t* wpb   = (bf16_t*)walloc((size_t)H_ * IN_ * 2);      // W_proj, bf16
  bf16_t* wqb   = (bf16_t*)walloc((size_t)H_ * H_ * 2);       // W_q, bf16
  bf16_t* wob   = (bf16_t*)walloc((size_t)H_ * H_ * 2);       // W_o, bf16
  bf16_t* wgb   = (bf16_t*)walloc((size_t)H_ * 2 * H_ * 2);   // W_g, bf16
  float*  xproj = (float* )walloc((size_t)B_ * H_ * 4);       // x@Wp^T+b, f32
  float*  qf    = (float* )walloc((size_t)B_ * H_ * 4);       // h@Wq^T+b, f32
  bf16_t* attnb = (bf16_t*)walloc((size_t)B_ * H_ * 2);       // attention out
  float*  ogf   = (float* )walloc((size_t)B_ * H_ * 4);       // gelu out, f32
  bf16_t* ogb   = (bf16_t*)walloc((size_t)B_ * H_ * 2);       // gelu out, bf16
  float*  gatef = (float* )walloc((size_t)B_ * H_ * 4);       // sigmoid gate

  // 1) f32 -> bf16 conversions
  cvt_f32_bf16<<<4096, 256, 0, stream>>>(x,      xb,  B_ * IN_);
  cvt_f32_bf16<<<4096, 256, 0, stream>>>(h_prev, hb,  B_ * H_);
  cvt_f32_bf16<<<2048, 256, 0, stream>>>(W_proj, wpb, H_ * IN_);
  cvt_f32_bf16<<<2048, 256, 0, stream>>>(W_q,    wqb, H_ * H_);
  cvt_f32_bf16<<<2048, 256, 0, stream>>>(W_o,    wob, H_ * H_);
  cvt_f32_bf16<<<4096, 256, 0, stream>>>(W_g,    wgb, H_ * 2 * H_);

  const dim3 gemm_grid(H_ / 128, B_ / 128);   // (16, 64)

  // 2) x_proj = x @ W_proj^T + b_proj
  gemm_bf16_wmma<0, IN_, IN_, IN_, IN_><<<gemm_grid, 256, 0, stream>>>(
      xb, xb, wpb, b_proj, xproj, (bf16_t*)nullptr);

  // 3) q = h_prev @ W_q^T + b_q
  gemm_bf16_wmma<0, H_, H_, H_, H_><<<gemm_grid, 256, 0, stream>>>(
      hb, hb, wqb, b_q, qf, (bf16_t*)nullptr);

  // 4) 4-key attention per (b, head) -> attnb (bf16)
  attn_kernel<<<B_, 128, 0, stream>>>(h_prev, xproj, qf, attnb);

  // 5) og = gelu(attn @ W_o^T + b_o)  (f32 for LN, bf16 for gate GEMM)
  gemm_bf16_wmma<1, H_, H_, H_, H_><<<gemm_grid, 256, 0, stream>>>(
      attnb, attnb, wob, b_o, ogf, ogb);

  // 6) gate = sigmoid([h_prev, og] @ W_g^T + b_g)   (K = 4096, split A)
  gemm_bf16_wmma<2, H_, 2 * H_, H_, H_><<<gemm_grid, 256, 0, stream>>>(
      hb, ogb, wgb, b_g, gatef, (bf16_t*)nullptr);

  // 7) layernorm(og + h_prev) gated with h_prev -> out
  final_kernel<<<B_, 256, 0, stream>>>(h_prev, ogf, gatef, gamma, beta, out);

  (void)in_sizes; (void)n_in; (void)out_size; (void)ws_size;
}
